// ImplicitNetworkGrid_COMBINE_22857815949563
// MI455X (gfx1250) — matrix operand order repaired and checked by reading, not compiled, not run on hardware
//
#include <hip/hip_runtime.h>

// ---------------- model constants ----------------
#define NUM_LEVELS 16
#define TABLE_SZ   (1 << 19)          // 524288 entries per level, 2 floats each
#define BPTS       128                // points per workgroup
#define H_DIM      256
#define CIN_PITCH  64                 // 35 valid, padded to 2 K-tiles of 32
#define FIN_PITCH  320                // 291 valid, padded to 10 K-tiles of 32
#define OUT_COLS   257
#define N_PTS      262144

typedef __attribute__((ext_vector_type(16))) _Float16 v16h;
typedef __attribute__((ext_vector_type(8)))  float    v8f;

// RES[l] = ceil(16 * 2^(7l/15)); dense if (res+1)^3 <= 2^19  (levels 0..4)
__constant__ int kRes[NUM_LEVELS] = {16, 23, 31, 43, 59, 81, 112, 154,
                                     213, 295, 407, 562, 777, 1073, 1483, 2048};

static __device__ __forceinline__ int imin(int a, int b) { return a < b ? a : b; }

// stable softplus100: logaddexp(0, 100x)/100
static __device__ __forceinline__ float softplus100(float v) {
    float t = 100.0f * v;
    float e = __expf(-fabsf(t));
    return (fmaxf(t, 0.0f) + __logf(1.0f + e)) * 0.01f;
}

// ---------------------------------------------------------------------------
// One MLP layer: [128 x K] (f16, LDS) x [K x nValid] (f32 weights W[n][k]) + b
// MODE 0: softplus -> f16 -> actOut LDS
// MODE 1: (coarse final) f32 -> Out[p][n]; cols 1.. -> f16 into f_in LDS (c_fv)
// MODE 2: (fine final)   Out[p][n] += acc  (same wave+lane wrote it in MODE 1)
// ---------------------------------------------------------------------------
template <int MODE>
static __device__ __forceinline__ void mlp_layer(
    const _Float16* __restrict__ actIn, int inPitch, int nKtiles, int kld,
    const float* __restrict__ W, const float* __restrict__ bias,
    int nValid, int nTiles,
    _Float16* __restrict__ actOut, int outPitch,
    float* __restrict__ gOut, _Float16* __restrict__ fFeed, int pBase)
{
    const int lane = threadIdx.x & 31;
    const int wave = threadIdx.x >> 5;
    const int lrow = lane & 15;   // A row / B col / D col within tile
    const int lhi  = lane >> 4;   // K-half select

    for (int nt = wave; nt < nTiles; nt += 8) {
        const int ncol = nt * 16 + lrow;
        const float bval = (ncol < nValid) ? bias[ncol] : 0.0f;

        v8f acc[8];
#pragma unroll
        for (int mt = 0; mt < 8; ++mt)
#pragma unroll
            for (int r = 0; r < 8; ++r) acc[mt][r] = bval;

        const float* wrow = W + (size_t)imin(ncol, nValid - 1) * kld;

        for (int kt = 0; kt < nKtiles; ++kt) {
            const int kbase = kt * 32;
            const int k = kbase + lhi * 16;

            // ---- B fragment: lane = column ncol, 16 consecutive k's ----
            float wv[16];
            if (ncol < nValid && (k + 16) <= kld) {
#pragma unroll
                for (int i = 0; i < 16; ++i) wv[i] = wrow[k + i];   // vectorizable
            } else {
#pragma unroll
                for (int i = 0; i < 16; ++i)
                    wv[i] = (ncol < nValid && (k + i) < kld) ? wrow[k + i] : 0.0f;
            }
            v16h bf;
#pragma unroll
            for (int i = 0; i < 16; ++i) bf[i] = (_Float16)wv[i];

            // ---- A fragments from LDS, 8 M-tiles reuse this B ----
#pragma unroll
            for (int mt = 0; mt < 8; ++mt) {
                const _Float16* ap =
                    actIn + (size_t)(mt * 16 + lrow) * inPitch + kbase + lhi * 16;
                union { v16h v; float4 f[2]; } au;
                au.f[0] = *(const float4*)(ap);        // K = kb+{0..7 | 16..23}
                au.f[1] = *(const float4*)(ap + 8);    // K = kb+{8..15 | 24..31}
                acc[mt] = __builtin_amdgcn_wmma_f32_16x16x32_f16(
                    false, au.v, false, bf, (short)0, acc[mt], false, false);
            }
        }

        // ---- epilogue: D layout VGPR r -> row r + 8*lhi, col = lrow ----
#pragma unroll
        for (int mt = 0; mt < 8; ++mt) {
#pragma unroll
            for (int r = 0; r < 8; ++r) {
                const int m = mt * 16 + r + lhi * 8;
                const float v = acc[mt][r];
                if (MODE == 0) {
                    actOut[(size_t)m * outPitch + ncol] = (_Float16)softplus100(v);
                } else if (MODE == 1) {
                    if (ncol < nValid) {
                        gOut[(size_t)(pBase + m) * OUT_COLS + ncol] = v;
                        if (ncol >= 1)
                            fFeed[(size_t)m * FIN_PITCH + 35 + (ncol - 1)] = (_Float16)v;
                    }
                } else {
                    if (ncol < nValid) {
                        float* op = gOut + (size_t)(pBase + m) * OUT_COLS + ncol;
                        *op = *op + v;  // same wave/lane wrote it: in-order per ISA
                    }
                }
            }
        }
    }
}

// ---------------------------------------------------------------------------
extern "C" __global__ void __launch_bounds__(256, 1)
ngp_fused_kernel(const float* __restrict__ X,
                 const float* __restrict__ cT, const float* __restrict__ cW0,
                 const float* __restrict__ cB0, const float* __restrict__ cW1,
                 const float* __restrict__ cB1, const float* __restrict__ cW2,
                 const float* __restrict__ cB2, const float* __restrict__ fT,
                 const float* __restrict__ fW0, const float* __restrict__ fB0,
                 const float* __restrict__ fW1, const float* __restrict__ fB1,
                 const float* __restrict__ fW2, const float* __restrict__ fB2,
                 float* __restrict__ Out)
{
    extern __shared__ _Float16 smem[];
    _Float16* sCin = smem;                        // 128 x 64
    _Float16* sFin = sCin + BPTS * CIN_PITCH;     // 128 x 320
    _Float16* sH1  = sFin + BPTS * FIN_PITCH;     // 128 x 256
    _Float16* sH2  = sH1  + BPTS * H_DIM;         // 128 x 256

    const int tid   = threadIdx.x;
    const int pBase = blockIdx.x * BPTS;
    const int p     = tid & (BPTS - 1);
    const int half  = tid >> 7;                   // levels [0,8) or [8,16)

    // =========================== hash encode ===========================
    const float* xg = X + (size_t)(pBase + p) * 3;
    float xr[3], xv[3];
#pragma unroll
    for (int d = 0; d < 3; ++d) {
        xr[d] = xg[d];
        float v = xr[d] * (1.0f / 1.5f);          // x / DIVIDE
        v = (v + 1.0f) * 0.5f;
        xv[d] = fminf(fmaxf(v, 0.0f), 1.0f);
    }

    if (half == 0) {
        // raw x into cols 0..2 of both inputs; zero the K pads
#pragma unroll
        for (int d = 0; d < 3; ++d) {
            sCin[(size_t)p * CIN_PITCH + d] = (_Float16)xr[d];
            sFin[(size_t)p * FIN_PITCH + d] = (_Float16)xr[d];
        }
        for (int j = 35; j < CIN_PITCH; ++j) sCin[(size_t)p * CIN_PITCH + j] = (_Float16)0.0f;
        for (int j = 291; j < FIN_PITCH; ++j) sFin[(size_t)p * FIN_PITCH + j] = (_Float16)0.0f;
    }

    for (int li = 0; li < 8; ++li) {
        const int l   = half * 8 + li;
        const int res = kRes[l];
        float w0[3]; int p0[3];
#pragma unroll
        for (int d = 0; d < 3; ++d) {
            float pp = xv[d] * (float)(res - 1);
            float fl = floorf(pp);
            p0[d] = (int)fl;
            w0[d] = pp - fl;
        }
        const float2* tc = (const float2*)cT + (size_t)l * TABLE_SZ;
        const float2* tf = (const float2*)fT + (size_t)l * TABLE_SZ;
        const bool dense = (res <= 79);           // (res+1)^3 <= 2^19

        float aC0 = 0.f, aC1 = 0.f, aF0 = 0.f, aF1 = 0.f;
#pragma unroll
        for (int c = 0; c < 8; ++c) {
            const int px = imin(p0[0] + (c & 1), res);
            const int py = imin(p0[1] + ((c >> 1) & 1), res);
            const int pz = imin(p0[2] + ((c >> 2) & 1), res);
            unsigned idx;
            if (dense) {
                idx = (unsigned)(px + (res + 1) * (py + (res + 1) * pz));
            } else {
                idx = ((unsigned)px ^ ((unsigned)py * 2654435761u)
                                   ^ ((unsigned)pz * 805459861u)) & (unsigned)(TABLE_SZ - 1);
            }
            const float wc = ((c & 1) ? w0[0] : 1.0f - w0[0]) *
                             ((c & 2) ? w0[1] : 1.0f - w0[1]) *
                             ((c & 4) ? w0[2] : 1.0f - w0[2]);
            const float2 vc = tc[idx];
            const float2 vf = tf[idx];
            aC0 += wc * vc.x; aC1 += wc * vc.y;
            aF0 += wc * vf.x; aF1 += wc * vf.y;
        }
        sCin[(size_t)p * CIN_PITCH + 3 + 2 * l]     = (_Float16)aC0;
        sCin[(size_t)p * CIN_PITCH + 3 + 2 * l + 1] = (_Float16)aC1;
        sFin[(size_t)p * FIN_PITCH + 3 + 2 * l]     = (_Float16)aF0;
        sFin[(size_t)p * FIN_PITCH + 3 + 2 * l + 1] = (_Float16)aF1;
    }
    __syncthreads();

    // =========================== coarse MLP ===========================
    mlp_layer<0>(sCin, CIN_PITCH, 2, 35,  cW0, cB0, 256, 16, sH1, H_DIM, nullptr, nullptr, pBase);
    __syncthreads();
    mlp_layer<0>(sH1, H_DIM, 8, 256, cW1, cB1, 256, 16, sH2, H_DIM, nullptr, nullptr, pBase);
    __syncthreads();
    mlp_layer<1>(sH2, H_DIM, 8, 256, cW2, cB2, 257, 17, nullptr, 0, Out, sFin, pBase);
    __syncthreads();

    // ============================ fine MLP ============================
    mlp_layer<0>(sFin, FIN_PITCH, 10, 291, fW0, fB0, 256, 16, sH1, H_DIM, nullptr, nullptr, pBase);
    __syncthreads();
    mlp_layer<0>(sH1, H_DIM, 8, 256, fW1, fB1, 256, 16, sH2, H_DIM, nullptr, nullptr, pBase);
    __syncthreads();
    mlp_layer<2>(sH2, H_DIM, 8, 256, fW2, fB2, 257, 17, nullptr, 0, Out, nullptr, pBase);
}

// ---------------------------------------------------------------------------
extern "C" void kernel_launch(void* const* d_in, const int* in_sizes, int n_in,
                              void* d_out, int out_size, void* d_ws, size_t ws_size,
                              hipStream_t stream)
{
    const float* X   = (const float*)d_in[0];
    const float* cT  = (const float*)d_in[1];
    const float* cW0 = (const float*)d_in[2];
    const float* cB0 = (const float*)d_in[3];
    const float* cW1 = (const float*)d_in[4];
    const float* cB1 = (const float*)d_in[5];
    const float* cW2 = (const float*)d_in[6];
    const float* cB2 = (const float*)d_in[7];
    const float* fT  = (const float*)d_in[8];
    const float* fW0 = (const float*)d_in[9];
    const float* fB0 = (const float*)d_in[10];
    const float* fW1 = (const float*)d_in[11];
    const float* fB1 = (const float*)d_in[12];
    const float* fW2 = (const float*)d_in[13];
    const float* fB2 = (const float*)d_in[14];
    float* Out = (float*)d_out;

    const size_t shmem =
        (size_t)BPTS * (CIN_PITCH + FIN_PITCH + 2 * H_DIM) * sizeof(_Float16); // 224 KB
    (void)hipFuncSetAttribute((const void*)ngp_fused_kernel,
                              hipFuncAttributeMaxDynamicSharedMemorySize, (int)shmem);

    hipLaunchKernelGGL(ngp_fused_kernel, dim3(N_PTS / BPTS), dim3(256), shmem, stream,
                       X, cT, cW0, cB0, cW1, cB1, cW2, cB2,
                       fT, fW0, fB0, fW1, fB1, fW2, fB2, Out);
}
